// my_SimCLR5_45561013076679
// MI455X (gfx1250) — matrix operand order repaired and checked by reading, not compiled, not run on hardware
//
#include <hip/hip_runtime.h>
#include <hip/hip_bf16.h>

// ---------------------------------------------------------------------------
// Shapes (from reference): B=64, A=312, DIM=2048, D=256, K=64, S=150
// Outputs (flat concat): logits_all [64*64] | part_CL [64*312*312] | cl_labels[64]
// ---------------------------------------------------------------------------
#define B_   64
#define A_   312
#define DIM_ 2048
#define D_   256
#define K_   64
#define S_   150
#define KPAD 320          // A padded to multiple of 32 for WMMA K-dim
#define NKS  (KPAD/32)    // 10 k-steps
#define NROW (B_*K_ + B_) // 4096 neg rows + 64 query rows = 4160
#define INV_T 8.3333333333333333f
#define SLOPE 0.01f

// workspace layout (float offsets; BMAT region is 32B aligned for v16bf loads)
#define WSUM_OFF   0                     // 2048 floats
#define BRSUM_OFF  2048                  // 1
#define COEF_OFF   2052                  // 64*64 = 4096
#define ATT_OFF    6148                  // 64*312 = 19968
#define BPROW_OFF  26116                 // 19968
#define CSUM_OFF   46084                 // 19968
#define BMAT_OFF   66056                 // 8*10*32*16 bf16 = 40960 bf16 = 20480 floats
#define Y_OFF      86536                 // 4160*128 = 532480 floats
// total ~619,016 floats (~2.48 MB)

typedef __attribute__((ext_vector_type(16))) __bf16 v16bf;
typedef __attribute__((ext_vector_type(8)))  __bf16 v8bf;
typedef __attribute__((ext_vector_type(8)))  float  v8f;

__device__ __forceinline__ float lrelu(float x) { return x > 0.f ? x : SLOPE * x; }

// ---------------------------------------------------------------------------
// K1: wsum[d] = sum_c Wr[c,d]  (Wr is [256,2048]); brsum = sum(br)
// ---------------------------------------------------------------------------
__global__ void k_wsum(const float* __restrict__ Wr, const float* __restrict__ br,
                       float* __restrict__ ws) {
  int d = blockIdx.x * 256 + threadIdx.x;        // 0..2047
  float acc = 0.f;
  for (int c = 0; c < D_; ++c) acc += Wr[c * DIM_ + d];   // coalesced across d
  ws[WSUM_OFF + d] = acc;
  if (d == 0) {
    float s = 0.f;
    for (int i = 0; i < D_; ++i) s += br[i];
    ws[BRSUM_OFF] = s;
  }
}

// ---------------------------------------------------------------------------
// K2: coef[b][b'] so that cache.sum(-1)[b] = sum_b' coef[b][b'] * bprow[b']
//     (closed form of the sequential EMA scan; ema starts at zero)
// ---------------------------------------------------------------------------
__global__ void k_coef(const int* __restrict__ labels, float* __restrict__ ws) {
  int b = threadIdx.x;                            // 64 threads
  int l = labels[b];
  for (int bp = 0; bp < B_; ++bp) {
    float w = 0.f;
    if (labels[bp] == l) {
      w = 0.8f;                                   // DECAY
      for (int j = bp + 1; j < B_; ++j)
        if (labels[j] == l) w *= 0.2f;            // (1-DECAY) per later overwrite
    }
    ws[COEF_OFF + b * B_ + bp] = w;
  }
}

// ---------------------------------------------------------------------------
// K3: att_proj[b,a] = leaky_relu(tar_atts[b] * W2[a] + b2[a])
// ---------------------------------------------------------------------------
__global__ void k_att(const float* __restrict__ tar, const float* __restrict__ W2,
                      const float* __restrict__ b2, float* __restrict__ ws) {
  int idx = blockIdx.x * 256 + threadIdx.x;       // 19968 threads exactly
  int b = idx / A_, a = idx % A_;
  ws[ATT_OFF + idx] = lrelu(tar[b] * W2[a] + b2[a]);
}

// ---------------------------------------------------------------------------
// K4: bprow[row] = part_feat[row,:] . wsum + brsum   (row = b*A + a)
//     One wave32 per 8KB row; float4 streaming + shfl_xor reduction.
//     Streams 166 MB -> ~7.2 us at 23.3 TB/s (the roofline floor).
// ---------------------------------------------------------------------------
__global__ void k_bprow(const float* __restrict__ pf, float* __restrict__ ws) {
  int lane = threadIdx.x & 31;
  int row  = blockIdx.x * 8 + (threadIdx.x >> 5); // 2496 blocks * 8 waves = 19968
  const float4* pf4 = (const float4*)pf + (size_t)row * (DIM_ / 4);
  const float4* ws4 = (const float4*)(ws + WSUM_OFF);
  float acc = 0.f;
  for (int i = 0; i < 16; ++i) {
    if (i + 2 < 16) __builtin_prefetch(&pf4[(i + 2) * 32 + lane], 0, 0);
    float4 x = pf4[i * 32 + lane];                // consecutive lanes: 16B coalesced
    float4 w = ws4[i * 32 + lane];                // L2-resident (8 KB)
    acc += x.x * w.x + x.y * w.y + x.z * w.z + x.w * w.w;
  }
  for (int off = 16; off > 0; off >>= 1) acc += __shfl_xor(acc, off);
  if (lane == 0) ws[BPROW_OFF + row] = acc + ws[BRSUM_OFF];
}

// ---------------------------------------------------------------------------
// K5: cache_sum[b,a] = sum_b' coef[b][b'] * bprow[b',a]
// ---------------------------------------------------------------------------
__global__ void k_csum(float* __restrict__ ws) {
  int idx = blockIdx.x * 256 + threadIdx.x;       // 19968
  int b = idx / A_, a = idx % A_;
  float acc = 0.f;
  for (int bp = 0; bp < B_; ++bp)
    acc += ws[COEF_OFF + b * B_ + bp] * ws[BPROW_OFF + bp * A_ + a];
  ws[CSUM_OFF + idx] = acc;
}

// ---------------------------------------------------------------------------
// K6: part_CL[b,i,k] = cache_sum[b,i] * bprow[b,k]  -> float4 stores (25 MB)
// ---------------------------------------------------------------------------
__global__ void k_outer(const float* __restrict__ ws, float* __restrict__ out) {
  int idx = blockIdx.x * 256 + threadIdx.x;       // 64*312*78 = 1,557,504 exactly
  int k4 = idx % 78;
  int i  = (idx / 78) % A_;
  int b  = idx / (78 * A_);
  float c = ws[CSUM_OFF + b * A_ + i];
  float4 r = *(const float4*)(ws + BPROW_OFF + b * A_ + k4 * 4);
  float4 o = make_float4(c * r.x, c * r.y, c * r.z, c * r.w);
  *(float4*)(out + B_ * K_ + ((size_t)(b * A_ + i)) * A_ + k4 * 4) = o;
}

// ---------------------------------------------------------------------------
// K7: Pre-swizzle W1 into the *per-lane WMMA B-fragment layout*:
//     Bfrag[nt][ks][lane][e]  (e innermost, contiguous 32B per lane)
//     element = B[k][n] = W1[n][k],  k = ks*32 + (lane>>4)*8 + (e&7) + (e>>3)*16,
//                                    n = nt*16 + (lane&15)
//     so the GEMM loads each fragment as one aligned v16bf (2x b128).
// ---------------------------------------------------------------------------
__global__ void k_bmat(const float* __restrict__ W1, float* __restrict__ ws) {
  int idx = blockIdx.x * 256 + threadIdx.x;       // 8*10*32*16 = 40960 exactly
  int e    = idx & 15;
  int lane = (idx >> 4) & 31;
  int t    = idx >> 9;                            // 0..79
  int ks   = t % NKS;
  int nt   = t / NKS;
  int k = ks * 32 + ((lane >> 4) << 3) + (e & 7) + ((e >> 3) << 4);
  int n = nt * 16 + (lane & 15);
  __bf16* Bf = (__bf16*)(ws + BMAT_OFF);
  Bf[idx] = (k < A_) ? (__bf16)W1[n * A_ + k] : (__bf16)0.f;
}

// ---------------------------------------------------------------------------
// K8: WMMA GEMM. Rows 0..4095: (v2s[sample_index[b,k]] + att[b]) @ W1^T
//     Rows 4096..4159 (query): (v2s[b] + att[b]) @ W1^T ;  then +b1, lrelu.
//     260 blocks * 128 threads (4 waves). Each wave: two 16x16 N-tiles,
//     10 k-steps of v_wmma_f32_16x16x32_bf16. A from LDS (2x ds_load_b128),
//     B from the pre-swizzled fragment buffer (2x global_load_b128).
// ---------------------------------------------------------------------------
__global__ void k_gemm(const float* __restrict__ v2s, const int* __restrict__ sidx,
                       const float* __restrict__ b1, float* __restrict__ ws) {
  __shared__ __attribute__((aligned(16))) __bf16 Alds[16 * KPAD];  // 10 KB
  int tid = threadIdx.x;
  int rt  = blockIdx.x;                           // row-tile, 260 tiles of 16

  // stage A tile (pad + gather + att-add fused)
  for (int idx = tid; idx < 16 * KPAD; idx += 128) {
    int m = idx / KPAD, a = idx % KPAD;
    int r = rt * 16 + m;
    float v = 0.f;
    if (a < A_) {
      if (r < B_ * K_) {
        int b = r >> 6, k = r & 63;
        int src = sidx[b * K_ + k];
        v = v2s[src * A_ + a] + ws[ATT_OFF + b * A_ + a];
      } else {
        int b = r - B_ * K_;
        v = v2s[b * A_ + a] + ws[ATT_OFF + b * A_ + a];
      }
    }
    Alds[m * KPAD + a] = (__bf16)v;
  }
  __syncthreads();

  const v16bf* Bf = (const v16bf*)(ws + BMAT_OFF);  // [nt][ks][lane] fragments
  int lane = tid & 31, wv = tid >> 5;
  int mrow = lane & 15;
  int hi   = lane >> 4;                           // lane half select
  int n0   = wv * 16 + (lane & 15);
  int n1   = (wv + 4) * 16 + (lane & 15);
  v8f acc0 = {}, acc1 = {};

  #pragma unroll
  for (int ks = 0; ks < NKS; ++ks) {
    // A-frag: lane holds row (lane&15); K = ks*32 + hi*8 + {0..7, 16..23}
    const __bf16* ap = &Alds[mrow * KPAD + ks * 32 + (hi << 3)];
    v8bf alo = *(const v8bf*)ap;                  // ds_load_b128
    v8bf ahi = *(const v8bf*)(ap + 16);           // ds_load_b128
    v16bf af = __builtin_shufflevector(alo, ahi, 0, 1, 2, 3, 4, 5, 6, 7,
                                       8, 9, 10, 11, 12, 13, 14, 15);
    // B-frags: one contiguous 32B fragment per lane (coalesced, L2-resident)
    v16bf bf0 = Bf[(wv       * NKS + ks) * 32 + lane];
    v16bf bf1 = Bf[((wv + 4) * NKS + ks) * 32 + lane];
    acc0 = __builtin_amdgcn_wmma_f32_16x16x32_bf16(false, af, false, bf0,
                                                   (short)0, acc0, false, false);
    acc1 = __builtin_amdgcn_wmma_f32_16x16x32_bf16(false, af, false, bf1,
                                                   (short)0, acc1, false, false);
  }

  // epilogue: D[r] -> row m = r + 8*hi, col n ; +b1, leaky_relu
  float* Y = ws + Y_OFF;
  float bn0 = b1[n0], bn1 = b1[n1];
  #pragma unroll
  for (int r = 0; r < 8; ++r) {
    int grow = rt * 16 + r + (hi << 3);
    Y[grow * 128 + n0] = lrelu(acc0[r] + bn0);
    Y[grow * 128 + n1] = lrelu(acc1[r] + bn1);
  }
}

// ---------------------------------------------------------------------------
// K9: logits[b,k] = (query[b] . pcn[b,k]) / T ; one wave32 per dot
// ---------------------------------------------------------------------------
__global__ void k_logits(const float* __restrict__ ws, float* __restrict__ out) {
  int lane = threadIdx.x & 31;
  int g    = blockIdx.x * 8 + (threadIdx.x >> 5); // 512 blocks * 8 = 4096
  int b = g >> 6, k = g & 63;
  const float* Y = ws + Y_OFF;
  const float* q = Y + (size_t)(B_ * K_ + b) * 128;
  const float* p = Y + (size_t)(b * K_ + k) * 128;
  float acc = 0.f;
  #pragma unroll
  for (int i = 0; i < 4; ++i) acc += q[lane + i * 32] * p[lane + i * 32];
  for (int off = 16; off > 0; off >>= 1) acc += __shfl_xor(acc, off);
  if (lane == 0) out[b * K_ + k] = acc * INV_T;
}

// ---------------------------------------------------------------------------
// K10: cl_labels = zeros (int32 zero == float zero bit pattern)
// ---------------------------------------------------------------------------
__global__ void k_tail(float* __restrict__ out) {
  out[B_ * K_ + (size_t)B_ * A_ * A_ + threadIdx.x] = 0.f;
}

extern "C" void kernel_launch(void* const* d_in, const int* in_sizes, int n_in,
                              void* d_out, int out_size, void* d_ws, size_t ws_size,
                              hipStream_t stream) {
  const float* v2s   = (const float*)d_in[0];
  const float* pf    = (const float*)d_in[1];
  const float* tar   = (const float*)d_in[2];
  // d_in[3] = ema (zeros; algebraically eliminated)
  const float* Wr    = (const float*)d_in[4];
  const float* br    = (const float*)d_in[5];
  const float* W1    = (const float*)d_in[6];
  const float* b1    = (const float*)d_in[7];
  const float* W2    = (const float*)d_in[8];
  const float* b2    = (const float*)d_in[9];
  const int*   labels = (const int*)d_in[10];
  const int*   sidx   = (const int*)d_in[11];
  float* out = (float*)d_out;
  float* ws  = (float*)d_ws;

  k_wsum  <<<8,    256, 0, stream>>>(Wr, br, ws);
  k_coef  <<<1,     64, 0, stream>>>(labels, ws);
  k_att   <<<78,   256, 0, stream>>>(tar, W2, b2, ws);
  k_bprow <<<2496, 256, 0, stream>>>(pf, ws);
  k_csum  <<<78,   256, 0, stream>>>(ws);
  k_outer <<<6084, 256, 0, stream>>>(ws, out);
  k_bmat  <<<160,  256, 0, stream>>>(W1, ws);
  k_gemm  <<<260,  128, 0, stream>>>(v2s, sidx, b1, ws);
  k_logits<<<512,  256, 0, stream>>>(ws, out);
  k_tail  <<<1,     64, 0, stream>>>(out);
}
